// MAST_38955353375011
// MI455X (gfx1250) — matrix operand order repaired and checked by reading, not compiled, not run on hardware
//
#include <hip/hip_runtime.h>
#include <cstdint>
#include <cstddef>

// ---------------------------------------------------------------------------
// MAST forward for MI455X (gfx1250): bf16 WMMA GEMMs + fp32 normalization.
// Compute-bound (~2.4 TFLOP vs ~1GB traffic @ 23.3TB/s) => everything big
// runs through v_wmma_f32_16x16x32_bf16 with fp32 accumulation, fed by
// double-buffered GLOBAL_LOAD_ASYNC_TO_LDS_B128 (ASYNCcnt) staging.
// Block tile 128x128, wave tile 32x64 => 8 WMMA per k-step per wave against
// 12 ds_load_b128 (2 A-frags + 4 B-frags).
// ---------------------------------------------------------------------------

typedef __attribute__((ext_vector_type(16))) __bf16 bf16x16;
typedef __attribute__((ext_vector_type(8)))  float  floatx8;

union FragCvt { uint4 u[2]; bf16x16 v; };

__device__ __forceinline__ float sigmoidf_(float x) { return 1.0f / (1.0f + expf(-x)); }

// Async copy 16B/lane from global (base + 32-bit offset) directly into LDS.
__device__ __forceinline__ void async_ld_b128(unsigned ldsOff, unsigned gOff,
                                              unsigned long long base)
{
    asm volatile("global_load_async_to_lds_b128 %0, %1, %2"
                 :: "v"(ldsOff), "v"(gOff), "s"(base) : "memory");
}

// ---------------- f32 -> bf16 conversion with optional zero column padding --
__global__ __launch_bounds__(256) void k_cvt_bf16(const float* __restrict__ src,
                                                  __bf16* __restrict__ dst,
                                                  long rows, int scols, int dcols)
{
    size_t idx = (size_t)blockIdx.x * 256 + threadIdx.x;
    size_t total = (size_t)rows * (size_t)dcols;
    if (idx >= total) return;
    size_t r = idx / (size_t)dcols;
    int    c = (int)(idx % (size_t)dcols);
    float v = (c < scols) ? src[r * (size_t)scols + c] : 0.0f;
    dst[idx] = (__bf16)v;
}

__global__ __launch_bounds__(256) void k_vadd(const float* __restrict__ a,
                                              const float* __restrict__ b,
                                              float* __restrict__ o, int n)
{
    int i = blockIdx.x * 256 + threadIdx.x;
    if (i < n) o[i] = a[i] + b[i];
}

// ---------------- WMMA GEMM: C = act(A * W^T + bias) -----------------------
// A: [M x K] bf16 row-major, W: [N x K] bf16 row-major (torch Linear weight).
// Block: 256 threads = 8 wave32s arranged 4(M) x 2(N); each wave owns a 32x64
// C sub-tile (2x4 WMMA 16x16 tiles). Block tile: 128 x 128, K-step 32.
// Global->LDS staging uses async-to-LDS loads with double-buffered LDS so the
// next K-slice streams in while the current one is multiplied.
__global__ __launch_bounds__(256) void k_gemm_wmma(
    const __bf16* __restrict__ A,
    const __bf16* __restrict__ W,
    const float*  __restrict__ bias,
    float*        __restrict__ Cf,
    __bf16*       __restrict__ Cb,
    int M, int N, int K, int relu)
{
    // +8 bf16 row pad: keeps 16B alignment of fragment chunks, spreads banks
    __shared__ __bf16 As[2][128][32 + 8];
    __shared__ __bf16 Bs[2][128][32 + 8];

    const int tid    = threadIdx.x;
    const int lane   = tid & 31;
    const int laneLo = lane & 15;
    const int laneHi = lane >> 4;
    const int wave   = tid >> 5;
    const int wm     = (wave & 3) * 32;   // 4 waves along M
    const int wn     = (wave >> 2) * 64;  // 2 waves along N
    const int bm     = blockIdx.x * 128;
    const int bn     = blockIdx.y * 128;

    // Per-thread staging slots: A tile = 512 x 16B (2/thread), B tile = 512 x 16B (2/thread).
    const int r0  = tid >> 2;          // 0..63
    const int r1  = r0 + 64;           // 64..127
    const int seg = tid & 3;           // 16B segment within 64B row slice
    const unsigned long long baseA = (unsigned long long)A;
    const unsigned long long baseW = (unsigned long long)W;
    const unsigned aoff0 = ((unsigned)(bm + r0) * (unsigned)K + (unsigned)seg * 8u) * 2u;
    const unsigned aoff1 = aoff0 + 64u * (unsigned)K * 2u;
    const unsigned boff0 = ((unsigned)(bn + r0) * (unsigned)K + (unsigned)seg * 8u) * 2u;
    const unsigned boff1 = boff0 + 64u * (unsigned)K * 2u;
    unsigned ldsA0[2], ldsA1[2], ldsB0[2], ldsB1[2];
    #pragma unroll
    for (int b = 0; b < 2; ++b) {
        ldsA0[b] = (unsigned)(size_t)&As[b][r0][seg * 8];
        ldsA1[b] = (unsigned)(size_t)&As[b][r1][seg * 8];
        ldsB0[b] = (unsigned)(size_t)&Bs[b][r0][seg * 8];
        ldsB1[b] = (unsigned)(size_t)&Bs[b][r1][seg * 8];
    }

    floatx8 acc[2][4] = {};

    const int ksteps = K >> 5;
    // Preload K-slice 0 into buffer 0 (4 async ops per wave).
    async_ld_b128(ldsA0[0], aoff0, baseA);
    async_ld_b128(ldsA1[0], aoff1, baseA);
    async_ld_b128(ldsB0[0], boff0, baseW);
    async_ld_b128(ldsB1[0], boff1, baseW);

    for (int ks = 0; ks < ksteps; ++ks) {
        const int buf = ks & 1;
        __syncthreads();   // all waves done reading buf^1 -> safe to refill it
        if (ks + 1 < ksteps) {
            const unsigned kb = (unsigned)(ks + 1) * 64u;
            async_ld_b128(ldsA0[buf ^ 1], aoff0 + kb, baseA);
            async_ld_b128(ldsA1[buf ^ 1], aoff1 + kb, baseA);
            async_ld_b128(ldsB0[buf ^ 1], boff0 + kb, baseW);
            async_ld_b128(ldsB1[buf ^ 1], boff1 + kb, baseW);
            // async loads complete in order: cnt<=4 => slice ks has landed
            asm volatile("s_wait_asynccnt 0x4" ::: "memory");
        } else {
            asm volatile("s_wait_asynccnt 0x0" ::: "memory");
        }
        __syncthreads();   // slice ks visible to every wave

        // CDNA5 16-bit A-matrix 16x32 layout: lanes 0-15 hold K {0..7,16..23},
        // lanes 16-31 hold K {8..15,24..31} of row m = laneLo.
        bf16x16 afrag[2], bfrag[4];
        #pragma unroll
        for (int i = 0; i < 2; ++i) {
            const int m = wm + i * 16 + laneLo;
            FragCvt f;
            f.u[0] = *reinterpret_cast<const uint4*>(&As[buf][m][laneHi * 8]);
            f.u[1] = *reinterpret_cast<const uint4*>(&As[buf][m][16 + laneHi * 8]);
            afrag[i] = f.v;
        }
        // B-matrix 32x16: lane column n = laneLo, K {0..7,16..23} (laneHi=0) / {8..15,24..31} (laneHi=1)
        #pragma unroll
        for (int j = 0; j < 4; ++j) {
            const int n = wn + j * 16 + laneLo;
            FragCvt f;
            f.u[0] = *reinterpret_cast<const uint4*>(&Bs[buf][n][laneHi * 8]);
            f.u[1] = *reinterpret_cast<const uint4*>(&Bs[buf][n][16 + laneHi * 8]);
            bfrag[j] = f.v;
        }
        #pragma unroll
        for (int i = 0; i < 2; ++i)
            #pragma unroll
            for (int j = 0; j < 4; ++j)
                acc[i][j] = __builtin_amdgcn_wmma_f32_16x16x32_bf16(
                    false, afrag[i], false, bfrag[j], (short)0, acc[i][j], false, false);
    }

    // C/D layout: element r -> (m = r + laneHi*8, n = laneLo)
    #pragma unroll
    for (int i = 0; i < 2; ++i) {
        #pragma unroll
        for (int j = 0; j < 4; ++j) {
            const int gn = bn + wn + j * 16 + laneLo;
            const float bv = bias ? bias[gn] : 0.0f;
            #pragma unroll
            for (int r = 0; r < 8; ++r) {
                const int gm = bm + wm + i * 16 + laneHi * 8 + r;
                float v = acc[i][j][r] + bv;
                if (relu) v = fmaxf(v, 0.0f);
                const size_t o = (size_t)gm * (size_t)N + gn;
                if (Cf) Cf[o] = v;
                if (Cb) Cb[o] = (__bf16)v;
            }
        }
    }
}

// ---------------- row layernorm (+optional residual, relu, f32/bf16 outs) ---
__global__ __launch_bounds__(256) void k_row_ln(
    const float* __restrict__ X, const float* __restrict__ R,
    const float* __restrict__ g, const float* __restrict__ b,
    float* __restrict__ outF, __bf16* __restrict__ outB, int D, int relu)
{
    __shared__ float red[256];
    __shared__ float s_mean, s_rstd;
    const size_t row = blockIdx.x;
    const float* x = X + row * (size_t)D;
    const float* r = R ? R + row * (size_t)D : nullptr;

    float sum = 0.0f;
    for (int i = threadIdx.x; i < D; i += 256) sum += x[i] + (r ? r[i] : 0.0f);
    red[threadIdx.x] = sum; __syncthreads();
    for (int s = 128; s > 0; s >>= 1) {
        if ((int)threadIdx.x < s) red[threadIdx.x] += red[threadIdx.x + s];
        __syncthreads();
    }
    if (threadIdx.x == 0) s_mean = red[0] / (float)D;
    __syncthreads();
    const float mean = s_mean;

    float vs = 0.0f;
    for (int i = threadIdx.x; i < D; i += 256) {
        float v = x[i] + (r ? r[i] : 0.0f) - mean; vs += v * v;
    }
    red[threadIdx.x] = vs; __syncthreads();
    for (int s = 128; s > 0; s >>= 1) {
        if ((int)threadIdx.x < s) red[threadIdx.x] += red[threadIdx.x + s];
        __syncthreads();
    }
    if (threadIdx.x == 0) s_rstd = rsqrtf(red[0] / (float)D + 1e-5f);
    __syncthreads();
    const float rstd = s_rstd;

    for (int i = threadIdx.x; i < D; i += 256) {
        float v = x[i] + (r ? r[i] : 0.0f);
        float o = (v - mean) * rstd * g[i] + b[i];
        if (relu) o = fmaxf(o, 0.0f);
        const size_t oi = row * (size_t)D + i;
        if (outF) outF[oi] = o;
        if (outB) outB[oi] = (__bf16)o;
    }
}

// ---------------- one-step bidirectional LSTM activation -------------------
// gates (i,f,g,o blocks of 1024): c = sig(i)*tanh(g); h = sig(o)*tanh(c)
__global__ __launch_bounds__(256) void k_lstm(const float* __restrict__ gf,
                                              const float* __restrict__ gb,
                                              __bf16* __restrict__ out, int M)
{
    size_t idx = (size_t)blockIdx.x * 256 + threadIdx.x;
    if (idx >= (size_t)M * 1024) return;
    size_t row = idx / 1024;
    int j = (int)(idx % 1024);
    {
        const float* G = gf + row * 4096;
        float c = sigmoidf_(G[j]) * tanhf(G[2048 + j]);
        out[row * 2048 + j] = (__bf16)(sigmoidf_(G[3072 + j]) * tanhf(c));
    }
    {
        const float* G = gb + row * 4096;
        float c = sigmoidf_(G[j]) * tanhf(G[2048 + j]);
        out[row * 2048 + 1024 + j] = (__bf16)(sigmoidf_(G[3072 + j]) * tanhf(c));
    }
}

// ---------------- small-sequence softmax attention -------------------------
// QKV: [nSeq*S x 3E] f32 (q|k|v), token row = n*S + s. One block per (n,h).
__global__ __launch_bounds__(256) void k_attention(
    const float* __restrict__ QKV, __bf16* __restrict__ out,
    int S, int heads, int D, int E)
{
    __shared__ float Ks[24 * 256];
    __shared__ float Vs[24 * 256];
    __shared__ float Sc[24 * 24 + 8];
    const int n = blockIdx.x / heads;
    const int h = blockIdx.x % heads;
    const size_t E3 = 3 * (size_t)E;
    const float scale = rsqrtf((float)D);

    for (int idx = threadIdx.x; idx < S * D; idx += 256) {
        int s = idx / D, d = idx % D;
        size_t ro = (size_t)(n * S + s) * E3 + (size_t)h * D + d;
        Ks[idx] = QKV[ro + E] * scale;
        Vs[idx] = QKV[ro + 2 * E];
    }
    __syncthreads();

    for (int p = threadIdx.x; p < S * S; p += 256) {
        int s = p / S, t = p % S;
        const float* q = QKV + (size_t)(n * S + s) * E3 + (size_t)h * D;
        float a = 0.0f;
        for (int d = 0; d < D; ++d) a += q[d] * Ks[t * D + d];
        Sc[p] = a;
    }
    __syncthreads();

    if ((int)threadIdx.x < S) {
        int s = threadIdx.x;
        float mx = -1e30f;
        for (int t = 0; t < S; ++t) mx = fmaxf(mx, Sc[s * S + t]);
        float sum = 0.0f;
        for (int t = 0; t < S; ++t) { float e = expf(Sc[s * S + t] - mx); Sc[s * S + t] = e; sum += e; }
        float inv = 1.0f / sum;
        for (int t = 0; t < S; ++t) Sc[s * S + t] *= inv;
    }
    __syncthreads();

    for (int idx = threadIdx.x; idx < S * D; idx += 256) {
        int s = idx / D, d = idx % D;
        float a = 0.0f;
        for (int t = 0; t < S; ++t) a += Sc[s * S + t] * Vs[t * D + d];
        out[(size_t)(n * S + s) * E + (size_t)h * D + d] = (__bf16)a;
    }
}

// ---------------- gather s == S-1 rows, f32 + bf16 outputs -----------------
__global__ __launch_bounds__(256) void k_gather_last(
    const float* __restrict__ t1, float* __restrict__ of, __bf16* __restrict__ ob,
    int S, int E, int nrows)
{
    size_t idx = (size_t)blockIdx.x * 256 + threadIdx.x;
    if (idx >= (size_t)nrows * E) return;
    size_t nr = idx / E;
    int e = (int)(idx % E);
    float v = t1[(nr * S + (S - 1)) * (size_t)E + e];
    of[idx] = v;
    ob[idx] = (__bf16)v;
}

// ---------------- mean over 24 tokens per batch ----------------------------
__global__ __launch_bounds__(256) void k_mean_pool(
    const float* __restrict__ t2, float* __restrict__ pooled, int S, int E)
{
    size_t idx = (size_t)blockIdx.x * 256 + threadIdx.x;
    if (idx >= (size_t)32 * E) return;
    size_t b = idx / E;
    int e = (int)(idx % E);
    float s = 0.0f;
    for (int c = 0; c < S; ++c) s += t2[(b * S + c) * (size_t)E + e];
    pooled[idx] = s / (float)S;
}

// ---------------- final dot: out[b] = pooled[b] . w + bias -----------------
__global__ __launch_bounds__(256) void k_fc_out(
    const float* __restrict__ pooled, const float* __restrict__ w,
    const float* __restrict__ bias, float* __restrict__ out, int E)
{
    __shared__ float red[256];
    const int b = blockIdx.x;
    float s = 0.0f;
    for (int i = threadIdx.x; i < E; i += 256) s += pooled[(size_t)b * E + i] * w[i];
    red[threadIdx.x] = s; __syncthreads();
    for (int k = 128; k > 0; k >>= 1) {
        if ((int)threadIdx.x < k) red[threadIdx.x] += red[threadIdx.x + k];
        __syncthreads();
    }
    if (threadIdx.x == 0) out[b] = red[0] + bias[0];
}

// ---------------------------------------------------------------------------
extern "C" void kernel_launch(void* const* d_in, const int* in_sizes, int n_in,
                              void* d_out, int out_size, void* d_ws, size_t ws_size,
                              hipStream_t stream)
{
    (void)in_sizes; (void)out_size; (void)ws_size;
    if (n_in < 105) return;

    // Input index map: jax.tree_util flatten order (dict keys sorted, lists in order).
    // params.attn1:  0 ff1.b  1 ff1.w  2 ff2.b  3 ff2.w  4 in_b  5 in_w
    //                6 ln1.b  7 ln1.g  8 ln2.b  9 ln2.g 10 out_b 11 out_w
    // params.attn2: 12..23 (same layout; ff dims 2048)
    // params.channel: 24 fc1.b 25 fc1.w 26 fc2.b 27 fc2.w 28 fc_input.b 29 fc_input.w
    //   lstm layer l @ 30+8l: +0 bwd.bhh +1 bwd.bih +2 bwd.whh +3 bwd.wih
    //                         +4 fwd.bhh +5 fwd.bih +6 fwd.whh +7 fwd.wih
    //   mlps blk/stage @ 54+4*(4*blk+st): +0 lin.b +1 lin.w +2 ln.b +3 ln.g
    // 102 fc_out.b  103 fc_out.w  104 x [32,24,16,2000]

    const int M  = 32 * 24 * 16;   // 12288 channel rows
    const int M2 = 32 * 24;        // 768 attn2 tokens
    const int E  = 2048;
    const int mdin[4]  = {2048, 1024, 512, 1024};
    const int mdout[4] = {1024, 512, 1024, 2048};

    char*  ws  = (char*)d_ws;
    size_t off = 0;
    auto alloc = [&](size_t bytes) -> void* {
        void* p = ws + off;
        off = (off + bytes + 255) & ~(size_t)255;
        return p;
    };
    auto F32 = [&](int i) { return (const float*)d_in[i]; };
    auto cvt = [&](const float* src, long rows, int scols, int dcols) -> __bf16* {
        __bf16* dst = (__bf16*)alloc((size_t)rows * dcols * sizeof(__bf16));
        size_t total = (size_t)rows * dcols;
        k_cvt_bf16<<<dim3((unsigned)((total + 255) / 256)), dim3(256), 0, stream>>>(
            src, dst, rows, scols, dcols);
        return dst;
    };
    auto gemm = [&](const __bf16* A, const __bf16* W, const float* bias,
                    float* Cf, __bf16* Cb, int m, int n, int k, int relu) {
        dim3 grid(m / 128, n / 128);
        k_gemm_wmma<<<grid, dim3(256), 0, stream>>>(A, W, bias, Cf, Cb, m, n, k, relu);
    };
    auto ln = [&](const float* X, const float* R, const float* g, const float* b,
                  float* outF, __bf16* outB, int rows, int D, int relu) {
        k_row_ln<<<dim3(rows), dim3(256), 0, stream>>>(X, R, g, b, outF, outB, D, relu);
    };

    // ---- weights -> bf16 (fc_input / x padded K: 2000 -> 2016 = 63*32) ----
    __bf16* Wfin = cvt(F32(29), 2048, 2000, 2016);
    __bf16* Wst[12];
    for (int s = 0; s < 12; ++s)
        Wst[s] = cvt(F32(54 + 4 * s + 1), mdout[s & 3], mdin[s & 3], mdin[s & 3]);
    __bf16 *WihF[3], *WihB[3];
    float  *BgF[3],  *BgB[3];
    for (int l = 0; l < 3; ++l) {
        int base = 30 + 8 * l;
        WihB[l] = cvt(F32(base + 3), 4096, 2048, 2048);
        WihF[l] = cvt(F32(base + 7), 4096, 2048, 2048);
        BgB[l] = (float*)alloc(4096 * sizeof(float));
        BgF[l] = (float*)alloc(4096 * sizeof(float));
        k_vadd<<<16, 256, 0, stream>>>(F32(base + 1), F32(base + 0), BgB[l], 4096);
        k_vadd<<<16, 256, 0, stream>>>(F32(base + 5), F32(base + 4), BgF[l], 4096);
    }
    __bf16* Wfc1  = cvt(F32(25), 1024, 2048, 2048);
    __bf16* Wfc2  = cvt(F32(27), 2048, 1024, 1024);
    __bf16* Win1  = cvt(F32(5),  6144, 2048, 2048);
    __bf16* Wout1 = cvt(F32(11), 2048, 2048, 2048);
    __bf16* Wf1a  = cvt(F32(1),  1024, 2048, 2048);
    __bf16* Wf2a  = cvt(F32(3),  2048, 1024, 1024);
    __bf16* Win2  = cvt(F32(17), 6144, 2048, 2048);
    __bf16* Wout2 = cvt(F32(23), 2048, 2048, 2048);
    __bf16* Wf1b  = cvt(F32(13), 2048, 2048, 2048);
    __bf16* Wf2b  = cvt(F32(15), 2048, 2048, 2048);

    // ---- activation buffers ----
    __bf16* Xb   = cvt(F32(104), M, 2000, 2016);
    __bf16* ACT0 = (__bf16*)alloc((size_t)M * E * 2);
    __bf16* ACT1 = (__bf16*)alloc((size_t)M * E * 2);
    float*  CHF  = (float*)alloc((size_t)M * E * 4);
    float*  X1F  = (float*)alloc((size_t)M * E * 4);
    float*  GF   = (float*)alloc((size_t)M * 8192 * 4);  // gemm f32 scratch / gates / qkv
    float*  GB2  = (float*)alloc((size_t)M * E * 4);
    float*  T2XF = (float*)alloc((size_t)M2 * E * 4);
    __bf16* T2XB = (__bf16*)alloc((size_t)M2 * E * 2);
    float*  X2F  = (float*)alloc((size_t)M2 * E * 4);
    __bf16* X2B  = (__bf16*)alloc((size_t)M2 * E * 2);
    __bf16* SB0  = (__bf16*)alloc((size_t)M2 * E * 2);
    float*  T2F  = (float*)alloc((size_t)M2 * E * 4);
    float*  POOL = (float*)alloc((size_t)32 * E * 4);

    // ---- channel: fc_input -> 12 MLP stages -> 3 biLSTM steps -> fc1 -> fc2
    gemm(Xb, Wfin, F32(28), nullptr, ACT0, M, 2048, 2016, 0);
    __bf16* cur = ACT0; __bf16* nxt = ACT1;
    for (int s = 0; s < 12; ++s) {
        int base = 54 + 4 * s;
        gemm(cur, Wst[s], F32(base + 0), GF, nullptr, M, mdout[s & 3], mdin[s & 3], 0);
        ln(GF, nullptr, F32(base + 3), F32(base + 2), nullptr, nxt, M, mdout[s & 3], 1);
        __bf16* t = cur; cur = nxt; nxt = t;
    }
    float* GATE_B = GF + (size_t)M * 4096;
    for (int l = 0; l < 3; ++l) {
        gemm(cur, WihF[l], BgF[l], GF,     nullptr, M, 4096, 2048, 0);
        gemm(cur, WihB[l], BgB[l], GATE_B, nullptr, M, 4096, 2048, 0);
        k_lstm<<<dim3((M * 1024 + 255) / 256), dim3(256), 0, stream>>>(GF, GATE_B, nxt, M);
        __bf16* t = cur; cur = nxt; nxt = t;
    }
    gemm(cur, Wfc1, F32(24), nullptr, nxt, M, 1024, 2048, 0);     // fc1 -> [M,1024] bf16
    gemm(nxt, Wfc2, F32(26), CHF, cur, M, 2048, 1024, 0);         // ch f32 + bf16
    __bf16* CHB = cur;
    __bf16* TMP = nxt;

    // ---- transformer 1: S=16 over inner slices, 16 heads, D=128 ----
    gemm(CHB, Win1, F32(4), GF, nullptr, M, 6144, 2048, 0);                       // qkv
    k_attention<<<dim3(768 * 16), dim3(256), 0, stream>>>(GF, TMP, 16, 16, 128, E);
    gemm(TMP, Wout1, F32(10), GB2, nullptr, M, 2048, 2048, 0);                    // out proj
    ln(GB2, CHF, F32(7), F32(6), X1F, TMP, M, 2048, 0);                           // ln1(x+mha)
    gemm(TMP, Wf1a, F32(0), nullptr, CHB, M, 1024, 2048, 1);                      // ff1+relu
    gemm(CHB, Wf2a, F32(2), GB2, nullptr, M, 2048, 1024, 0);                      // ff2
    float* T1F = GF;                                                              // reuse qkv buf
    ln(GB2, X1F, F32(9), F32(8), T1F, nullptr, M, 2048, 0);                       // ln2(x+ff)

    // ---- last token of each 16-seq -> [768, E] ----
    k_gather_last<<<dim3((M2 * E + 255) / 256), dim3(256), 0, stream>>>(
        T1F, T2XF, T2XB, 16, E, M2);

    // ---- transformer 2: S=24 over channels, 8 heads, D=256 ----
    float* QKV2 = GF + (size_t)M * 2048;     // disjoint from T1F region
    gemm(T2XB, Win2, F32(16), QKV2, nullptr, M2, 6144, 2048, 0);
    k_attention<<<dim3(32 * 8), dim3(256), 0, stream>>>(QKV2, SB0, 24, 8, 256, E);
    gemm(SB0, Wout2, F32(22), GB2, nullptr, M2, 2048, 2048, 0);
    ln(GB2, T2XF, F32(19), F32(18), X2F, X2B, M2, 2048, 0);
    gemm(X2B, Wf1b, F32(12), nullptr, SB0, M2, 2048, 2048, 1);
    gemm(SB0, Wf2b, F32(14), GB2, nullptr, M2, 2048, 2048, 0);
    ln(GB2, X2F, F32(21), F32(20), T2F, nullptr, M2, 2048, 0);

    // ---- mean pool over 24 tokens + fc_out ----
    k_mean_pool<<<dim3((32 * E + 255) / 256), dim3(256), 0, stream>>>(T2F, POOL, 24, E);
    k_fc_out<<<dim3(32), dim3(256), 0, stream>>>(POOL, F32(103), F32(102), (float*)d_out, E);
}
